// RGAT_46746424049911
// MI455X (gfx1250) — matrix operand order repaired and checked by reading, not compiled, not run on hardware
//
#include <hip/hip_runtime.h>
#include <hip/hip_bf16.h>

#define N_NODES 50000
#define N_EDGES 800000
#define IN_FEAT 128
#define HIDDEN 128
#define HEAD_DIM 64
#define N_CLASSES 16
#define NEG_SLOPE 0.2f
#define BN_EPS 1e-5f

typedef __attribute__((ext_vector_type(16))) __bf16 v16bf;
typedef __attribute__((ext_vector_type(8)))  float v8f;

// ---------- helpers ----------
__device__ __forceinline__ unsigned fkey(float f) {
    unsigned b = __float_as_uint(f);
    return (b & 0x80000000u) ? ~b : (b | 0x80000000u);
}
__device__ __forceinline__ float funkey(unsigned k) {
    return (k & 0x80000000u) ? __uint_as_float(k & 0x7fffffffu) : __uint_as_float(~k);
}
#define NEG_INF_KEY 0x007FFFFFu  // fkey(-inf)

// ---------- tiny utility kernels ----------
__global__ void rgat_cvt_bf16(const float* __restrict__ in, __bf16* __restrict__ out, int n) {
    int t = blockIdx.x * blockDim.x + threadIdx.x;
    if (t < n) out[t] = (__bf16)in[t];
}

__global__ void rgat_fill_u32(unsigned* __restrict__ p, unsigned v, int n) {
    int t = blockIdx.x * blockDim.x + threadIdx.x;
    if (t < n) p[t] = v;
}

__global__ void rgat_h2_init(float* __restrict__ h2, const float* __restrict__ b0,
                             const float* __restrict__ b1, int n) {
    int t = blockIdx.x * blockDim.x + threadIdx.x;
    if (t < n) { int c = t & 127; h2[t] = b0[c] + b1[c]; }
}

// ---------- fused 3-matrix WMMA GEMM: F = X @ W^T  (X:[N,128]bf16, W:[128,128]bf16) ----------
// grid = (N/16, 3), block = 32 (one wave per 16-row tile of one output matrix)
__global__ __launch_bounds__(32) void rgat_wmma_gemm3(
    const __bf16* __restrict__ xb, const __bf16* __restrict__ wb,
    float* __restrict__ F0, float* __restrict__ F1, float* __restrict__ S)
{
    int lane = threadIdx.x & 31;
    int mat  = blockIdx.y;
    const __bf16* W = wb + (size_t)mat * HIDDEN * IN_FEAT;
    float* OUT = (mat == 0) ? F0 : (mat == 1 ? F1 : S);

    int r0   = blockIdx.x * 16;
    int mrow = r0 + (lane & 15);
    int aoff = (lane < 16) ? 0 : 8;    // A: 16-bit 16x32 layout (ISA 7.12.2)
    int koff = (lane < 16) ? 0 : 8;    // B: same 16-bit layout as A (K 8h..8h+7, 16+8h..16+8h+7)

    // Load A fragments once (K = 128 -> 4 chunks of 32)
    v16bf a[4];
#pragma unroll
    for (int kc = 0; kc < 4; ++kc) {
        const __bf16* xr = xb + (size_t)mrow * IN_FEAT + kc * 32;
#pragma unroll
        for (int e = 0; e < 8; ++e) {
            a[kc][e]     = xr[aoff + e];
            a[kc][8 + e] = xr[16 + aoff + e];
        }
    }

    for (int ct = 0; ct < 8; ++ct) {
        int n = ct * 16 + (lane & 15);
        v8f acc = {};
#pragma unroll
        for (int kc = 0; kc < 4; ++kc) {
            const __bf16* wr = W + (size_t)n * IN_FEAT + kc * 32 + koff;
            v16bf b;
#pragma unroll
            for (int e = 0; e < 16; ++e) b[e] = wr[e < 8 ? e : e + 8];
            acc = __builtin_amdgcn_wmma_f32_16x16x32_bf16(
                false, a[kc], false, b, (short)0, acc, false, false);
        }
        int mbase = r0 + ((lane < 16) ? 0 : 8);
#pragma unroll
        for (int g = 0; g < 8; ++g)
            OUT[(size_t)(mbase + g) * HIDDEN + ct * 16 + (lane & 15)] = acc[g];
    }
}

// ---------- attention projections: el/er per (node, head) ----------
__global__ void rgat_el_er(const float* __restrict__ F, const float* __restrict__ al,
                           const float* __restrict__ ar, float* __restrict__ el,
                           float* __restrict__ er, int n2)
{
    int t = blockIdx.x * blockDim.x + threadIdx.x;
    if (t >= n2) return;
    int i = t >> 1, h = t & 1;
    const float* f = F + (size_t)i * HIDDEN + h * HEAD_DIM;
    const float* l = al + h * HEAD_DIM;
    const float* r = ar + h * HEAD_DIM;
    float sl = 0.f, sr = 0.f;
#pragma unroll 4
    for (int d = 0; d < HEAD_DIM; ++d) { float v = f[d]; sl += v * l[d]; sr += v * r[d]; }
    el[t] = sl; er[t] = sr;
}

// ---------- edge pass 1: keyed float atomicMax per (dst, head) ----------
__global__ void rgat_edge_max(const int* __restrict__ src, const int* __restrict__ dst,
                              const float* __restrict__ el, const float* __restrict__ er,
                              unsigned* __restrict__ mkey, int n)
{
    int t = blockIdx.x * blockDim.x + threadIdx.x;
    if (t >= n) return;
    int e = t >> 1, h = t & 1;
    int s = src[e], d = dst[e];
    float v = el[s * 2 + h] + er[d * 2 + h];
    v = v > 0.f ? v : NEG_SLOPE * v;
    atomicMax(&mkey[d * 2 + h], fkey(v));
}

// ---------- edge pass 2: exp(e - max), denominator atomicAdd ----------
__global__ void rgat_edge_exp(const int* __restrict__ src, const int* __restrict__ dst,
                              const float* __restrict__ el, const float* __restrict__ er,
                              const unsigned* __restrict__ mkey, float* __restrict__ exb,
                              float* __restrict__ den, int n)
{
    int t = blockIdx.x * blockDim.x + threadIdx.x;
    if (t >= n) return;
    int e = t >> 1, h = t & 1;
    int s = src[e], d = dst[e];
    float v = el[s * 2 + h] + er[d * 2 + h];
    v = v > 0.f ? v : NEG_SLOPE * v;
    float ex = __expf(v - funkey(mkey[d * 2 + h]));
    exb[t] = ex;
    atomicAdd(&den[d * 2 + h], ex);
}

// ---------- edge pass 3: alpha * f[src] scatter-add (float4 gather, f32 atomics in L2) ----------
__global__ __launch_bounds__(256) void rgat_edge_agg(
    const int* __restrict__ src, const int* __restrict__ dst, const float* __restrict__ ew,
    const float* __restrict__ exb, const float* __restrict__ den,
    const float* __restrict__ F, float* __restrict__ h2, long long total)
{
    long long t = (long long)blockIdx.x * 256 + threadIdx.x;
    if (t >= total) return;
    int e = (int)(t >> 5);
    int q = (int)(t & 31);
    int c = q * 4;
    int h = c >> 6;
    int s = src[e], d = dst[e];
    float alpha = exb[e * 2 + h] / den[d * 2 + h] * ew[e];
    const float4 fv = *(const float4*)(F + (size_t)s * HIDDEN + c);
    float* o = h2 + (size_t)d * HIDDEN + c;
    atomicAdd(o + 0, alpha * fv.x);
    atomicAdd(o + 1, alpha * fv.y);
    atomicAdd(o + 2, alpha * fv.z);
    atomicAdd(o + 3, alpha * fv.w);
}

// ---------- fused: h = S + skip_b + relu(h2); per-channel BN partial sums ----------
__global__ __launch_bounds__(128) void rgat_bn_reduce(
    const float* __restrict__ S, const float* __restrict__ h2, const float* __restrict__ skip_b,
    float* __restrict__ hout, float* __restrict__ bnsum, float* __restrict__ bnsumsq,
    int rows_per_block, int nrows)
{
    int c = threadIdx.x;
    int r0 = blockIdx.x * rows_per_block;
    int r1 = r0 + rows_per_block; if (r1 > nrows) r1 = nrows;
    float sb = skip_b[c];
    float s = 0.f, ss = 0.f;
    for (int r = r0; r < r1; ++r) {
        float g = h2[(size_t)r * HIDDEN + c];
        float v = S[(size_t)r * HIDDEN + c] + sb + (g > 0.f ? g : 0.f);
        hout[(size_t)r * HIDDEN + c] = v;
        s += v; ss += v * v;
    }
    atomicAdd(&bnsum[c], s);
    atomicAdd(&bnsumsq[c], ss);
}

__global__ void rgat_bn_finalize(const float* __restrict__ bnsum, const float* __restrict__ bnsumsq,
                                 const float* __restrict__ gamma, const float* __restrict__ beta,
                                 float* __restrict__ scale, float* __restrict__ shift, float inv_n)
{
    int c = threadIdx.x;
    float mu  = bnsum[c] * inv_n;
    float var = bnsumsq[c] * inv_n - mu * mu;
    float sc  = gamma[c] * rsqrtf(var + BN_EPS);
    scale[c] = sc;
    shift[c] = beta[c] - mu * sc;
}

// ---------- WMMA classifier: logits = relu(bn(h)) @ clf_w^T + clf_b ----------
// grid = N/16, block = 32: one wave -> one 16x16 output tile (16 classes)
__global__ __launch_bounds__(32) void rgat_wmma_clf(
    const float* __restrict__ h, const __bf16* __restrict__ clfwb,
    const float* __restrict__ scale, const float* __restrict__ shift,
    const float* __restrict__ clf_b, float* __restrict__ out)
{
    int lane = threadIdx.x & 31;
    int r0   = blockIdx.x * 16;
    int row  = r0 + (lane & 15);
    int aoff = (lane < 16) ? 0 : 8;
    int koff = (lane < 16) ? 0 : 8;
    int n    = lane & 15;

    v8f acc;
    float cb = clf_b[n];
#pragma unroll
    for (int g = 0; g < 8; ++g) acc[g] = cb;

#pragma unroll
    for (int kc = 0; kc < 4; ++kc) {
        int kb = kc * 32;
        const float* hr = h + (size_t)row * HIDDEN + kb;
        v16bf a, b;
#pragma unroll
        for (int e = 0; e < 8; ++e) {
            int k0 = kb + aoff + e;
            int k1 = kb + 16 + aoff + e;
            float v0 = hr[aoff + e]      * scale[k0] + shift[k0];
            float v1 = hr[16 + aoff + e] * scale[k1] + shift[k1];
            a[e]     = (__bf16)(v0 > 0.f ? v0 : 0.f);
            a[8 + e] = (__bf16)(v1 > 0.f ? v1 : 0.f);
        }
        const __bf16* wr = clfwb + (size_t)n * HIDDEN + kb + koff;
#pragma unroll
        for (int e = 0; e < 16; ++e) b[e] = wr[e < 8 ? e : e + 8];
        acc = __builtin_amdgcn_wmma_f32_16x16x32_bf16(
            false, a, false, b, (short)0, acc, false, false);
    }
    int mbase = r0 + ((lane < 16) ? 0 : 8);
#pragma unroll
    for (int g = 0; g < 8; ++g)
        out[(size_t)(mbase + g) * N_CLASSES + n] = acc[g];
}

// ---------- host launcher ----------
extern "C" void kernel_launch(void* const* d_in, const int* in_sizes, int n_in,
                              void* d_out, int out_size, void* d_ws, size_t ws_size,
                              hipStream_t stream) {
    const float* x       = (const float*)d_in[0];
    const int*   src0    = (const int*)  d_in[1];
    const int*   dst0    = (const int*)  d_in[2];
    const float* ew0     = (const float*)d_in[3];
    const int*   src1    = (const int*)  d_in[4];
    const int*   dst1    = (const int*)  d_in[5];
    const float* ew1     = (const float*)d_in[6];
    const float* fc_w0   = (const float*)d_in[7];
    const float* bias0   = (const float*)d_in[8];
    const float* attn_l0 = (const float*)d_in[9];
    const float* attn_r0 = (const float*)d_in[10];
    const float* fc_w1   = (const float*)d_in[11];
    const float* bias1   = (const float*)d_in[12];
    const float* attn_l1 = (const float*)d_in[13];
    const float* attn_r1 = (const float*)d_in[14];
    const float* skip_w  = (const float*)d_in[15];
    const float* skip_b  = (const float*)d_in[16];
    const float* bn_g    = (const float*)d_in[17];
    const float* bn_b    = (const float*)d_in[18];
    const float* clf_w   = (const float*)d_in[19];
    const float* clf_b   = (const float*)d_in[20];
    float* out = (float*)d_out;

    // workspace carve-up
    char* ws = (char*)d_ws;
    size_t off = 0;
    auto carve = [&](size_t bytes) { void* p = ws + off; off += (bytes + 255) & ~(size_t)255; return p; };
    __bf16* XB    = (__bf16*)carve((size_t)N_NODES * IN_FEAT * 2);
    __bf16* WB    = (__bf16*)carve((size_t)3 * HIDDEN * IN_FEAT * 2);
    __bf16* CLFWB = (__bf16*)carve((size_t)N_CLASSES * HIDDEN * 2);
    float*  F0    = (float*) carve((size_t)N_NODES * HIDDEN * 4);
    float*  F1    = (float*) carve((size_t)N_NODES * HIDDEN * 4);
    float*  S     = (float*) carve((size_t)N_NODES * HIDDEN * 4);
    float*  H2    = (float*) carve((size_t)N_NODES * HIDDEN * 4);
    float*  EL0   = (float*) carve((size_t)N_NODES * 2 * 4);
    float*  ER0   = (float*) carve((size_t)N_NODES * 2 * 4);
    float*  EL1   = (float*) carve((size_t)N_NODES * 2 * 4);
    float*  ER1   = (float*) carve((size_t)N_NODES * 2 * 4);
    unsigned* M0  = (unsigned*)carve((size_t)N_NODES * 2 * 4);
    unsigned* M1  = (unsigned*)carve((size_t)N_NODES * 2 * 4);
    float*  DEN0  = (float*) carve((size_t)N_NODES * 2 * 4);
    float*  DEN1  = (float*) carve((size_t)N_NODES * 2 * 4);
    float*  EX0   = (float*) carve((size_t)N_EDGES * 2 * 4);
    float*  EX1   = (float*) carve((size_t)N_EDGES * 2 * 4);
    float*  BNSUM = (float*) carve(HIDDEN * 4);
    float*  BNSSQ = (float*) carve(HIDDEN * 4);
    float*  SCALE = (float*) carve(HIDDEN * 4);
    float*  SHIFT = (float*) carve(HIDDEN * 4);
    (void)ws_size; (void)n_in; (void)in_sizes; (void)out_size;

    const int B = 256;
    // 1) bf16 conversions
    rgat_cvt_bf16<<<(N_NODES * IN_FEAT + B - 1) / B, B, 0, stream>>>(x, XB, N_NODES * IN_FEAT);
    rgat_cvt_bf16<<<(HIDDEN * IN_FEAT + B - 1) / B, B, 0, stream>>>(fc_w0,  WB,                      HIDDEN * IN_FEAT);
    rgat_cvt_bf16<<<(HIDDEN * IN_FEAT + B - 1) / B, B, 0, stream>>>(fc_w1,  WB + HIDDEN * IN_FEAT,   HIDDEN * IN_FEAT);
    rgat_cvt_bf16<<<(HIDDEN * IN_FEAT + B - 1) / B, B, 0, stream>>>(skip_w, WB + 2 * HIDDEN * IN_FEAT, HIDDEN * IN_FEAT);
    rgat_cvt_bf16<<<(N_CLASSES * HIDDEN + B - 1) / B, B, 0, stream>>>(clf_w, CLFWB, N_CLASSES * HIDDEN);

    // 2) three feature GEMMs via WMMA
    rgat_wmma_gemm3<<<dim3(N_NODES / 16, 3), 32, 0, stream>>>(XB, WB, F0, F1, S);

    // 3) attention projections
    rgat_el_er<<<(N_NODES * 2 + B - 1) / B, B, 0, stream>>>(F0, attn_l0, attn_r0, EL0, ER0, N_NODES * 2);
    rgat_el_er<<<(N_NODES * 2 + B - 1) / B, B, 0, stream>>>(F1, attn_l1, attn_r1, EL1, ER1, N_NODES * 2);

    // 4) init reduction buffers (every call; deterministic under graph replay)
    rgat_fill_u32<<<(N_NODES * 2 + B - 1) / B, B, 0, stream>>>(M0, NEG_INF_KEY, N_NODES * 2);
    rgat_fill_u32<<<(N_NODES * 2 + B - 1) / B, B, 0, stream>>>(M1, NEG_INF_KEY, N_NODES * 2);
    rgat_fill_u32<<<(N_NODES * 2 + B - 1) / B, B, 0, stream>>>((unsigned*)DEN0, 0u, N_NODES * 2);
    rgat_fill_u32<<<(N_NODES * 2 + B - 1) / B, B, 0, stream>>>((unsigned*)DEN1, 0u, N_NODES * 2);
    rgat_fill_u32<<<1, HIDDEN, 0, stream>>>((unsigned*)BNSUM, 0u, HIDDEN);
    rgat_fill_u32<<<1, HIDDEN, 0, stream>>>((unsigned*)BNSSQ, 0u, HIDDEN);
    rgat_h2_init<<<(N_NODES * HIDDEN + B - 1) / B, B, 0, stream>>>(H2, bias0, bias1, N_NODES * HIDDEN);

    // 5) edge softmax + weighted aggregation, both relations
    int e2 = N_EDGES * 2;
    rgat_edge_max<<<(e2 + B - 1) / B, B, 0, stream>>>(src0, dst0, EL0, ER0, M0, e2);
    rgat_edge_max<<<(e2 + B - 1) / B, B, 0, stream>>>(src1, dst1, EL1, ER1, M1, e2);
    rgat_edge_exp<<<(e2 + B - 1) / B, B, 0, stream>>>(src0, dst0, EL0, ER0, M0, EX0, DEN0, e2);
    rgat_edge_exp<<<(e2 + B - 1) / B, B, 0, stream>>>(src1, dst1, EL1, ER1, M1, EX1, DEN1, e2);
    long long aggtot = (long long)N_EDGES * 32;
    unsigned aggblocks = (unsigned)((aggtot + B - 1) / B);
    rgat_edge_agg<<<aggblocks, B, 0, stream>>>(src0, dst0, ew0, EX0, DEN0, F0, H2, aggtot);
    rgat_edge_agg<<<aggblocks, B, 0, stream>>>(src1, dst1, ew1, EX1, DEN1, F1, H2, aggtot);

    // 6) h = skip + relu(h2), BN stats (h written into F0, which is now free)
    int rpb = 64;
    rgat_bn_reduce<<<(N_NODES + rpb - 1) / rpb, HIDDEN, 0, stream>>>(S, H2, skip_b, F0, BNSUM, BNSSQ, rpb, N_NODES);
    rgat_bn_finalize<<<1, HIDDEN, 0, stream>>>(BNSUM, BNSSQ, bn_g, bn_b, SCALE, SHIFT, 1.0f / N_NODES);

    // 7) classifier via WMMA
    rgat_wmma_clf<<<N_NODES / 16, 32, 0, stream>>>(F0, CLFWB, SCALE, SHIFT, clf_b, out);
}
